// Better_Transformer_75634374082794
// MI455X (gfx1250) — matrix operand order repaired and checked by reading, not compiled, not run on hardware
//
#include <hip/hip_runtime.h>
#include <hip/hip_bf16.h>

typedef __bf16 bf16_t;
typedef __attribute__((ext_vector_type(16))) __bf16 v16bf;
typedef __attribute__((ext_vector_type(8)))  __bf16 v8bf;
typedef __attribute__((ext_vector_type(4)))  __bf16 v4bf;
typedef __attribute__((ext_vector_type(8)))  float  v8f;

#define B_ROWS   16384
#define IN_SIZE  4096
#define NPART    8
#define DDIM     512
#define ROWS     64          // rows per workgroup tile
#define MTILES   4           // 16-row M tiles per wave (64 rows)
#define APAD     520         // bf16 row stride (padded: 1040B -> conflict-free ds_b128)
#define KT_STEPS (DDIM / 32) // 16 K-steps of the 16x16x32 WMMA

// ---------------------------------------------------------------------------
// Weight prep: fp32 row-major W[p][k][n] -> bf16 in a contiguous-k B-fragment
// lane layout for v_wmma_f32_16x16x32_bf16 (same k order as load_a_frag):
//   fragment (p, ntile, ktile): lane = (kr>>4)*16 + (n&15) holds 16 contiguous
//   bf16 (j = kr&15), i.e. each lane's fragment slice is a contiguous 32B run.
// ---------------------------------------------------------------------------
__global__ __launch_bounds__(256)
void bt_prep_weights(const float* __restrict__ w1, const float* __restrict__ w2,
                     bf16_t* __restrict__ wf) {
  const unsigned tid   = blockIdx.x * 256u + threadIdx.x;   // 2 * 8 * 512 * 512 total
  const unsigned which = tid >> 21;                         // 0 = w1, 1 = w2
  const unsigned r     = tid & 0x1FFFFFu;                   // p*D*D + k*D + n
  const unsigned p     = r >> 18;
  const unsigned rem   = r & 0x3FFFFu;
  const unsigned k     = rem >> 9;
  const unsigned n     = rem & 511u;
  const float v = (which ? w2 : w1)[r];
  const unsigned nt = n >> 4;
  const unsigned nr = n & 15u;
  const unsigned kt = k >> 5;
  const unsigned kr = k & 31u;
  const unsigned lane = ((kr >> 4) << 4) | nr;
  const unsigned j    = kr & 15u;
  const size_t off = (size_t)which * (NPART * DDIM * DDIM)
                   + ((size_t)(p * 32u + nt) * 16u + kt) * 512u + lane * 16u + j;
  wf[off] = (bf16_t)v;
}

// Build a 16x32 bf16 A-fragment from a row-major LDS tile.
// Layout: lanes 0-15 (row M = lane) hold K = k0..k0+15;
// lanes 16-31 hold K = k0+16..k0+31  (handled via hi*16 offset).
__device__ __forceinline__ v16bf load_a_frag(const bf16_t* rowptr, int k0) {
  v8bf lo = *(const v8bf*)(rowptr + k0);
  v8bf hh = *(const v8bf*)(rowptr + k0 + 8);
  return __builtin_shufflevector(lo, hh, 0, 1, 2, 3, 4, 5, 6, 7,
                                         8, 9, 10, 11, 12, 13, 14, 15);
}

// Fast parametric swish: (g + sigmoid(bb*v)*(1-g)) * v
// v_exp_f32 + v_rcp_f32 instead of the IEEE division expansion.
__device__ __forceinline__ float pswish(float v, float g, float bb) {
  const float e  = __expf(-bb * v);
  const float sg = __builtin_amdgcn_rcpf(1.0f + e);
  return (g + sg * (1.0f - g)) * v;
}

// 64x512 tile times 512x512 weight block; each wave owns a 64x64 slab
// (4 M-tiles x 4 N-tiles of 16x16 f32 accumulators = 128 acc VGPRs).
// Per kt: 8 ds_load_b128 (A) + 8 global_load_b128 (B) feed 16 v_wmma.
__device__ __forceinline__ void gemm_tile(const bf16_t (*tile)[APAD],
                                          const bf16_t* __restrict__ wbase,
                                          int wave, int lane, v8f acc[MTILES][4]) {
  const int lr = lane & 15;
  const int hi = lane >> 4;
  for (int kt = 0; kt < KT_STEPS; ++kt) {
    const int k0 = kt * 32 + hi * 16;
    v16bf a[MTILES];
#pragma unroll
    for (int mt = 0; mt < MTILES; ++mt)
      a[mt] = load_a_frag(&tile[mt * 16 + lr][0], k0);
#pragma unroll
    for (int i = 0; i < 4; ++i) {
      const int nt = wave * 4 + i;
      const bf16_t* bp = wbase + ((size_t)(nt * 16 + kt)) * 512 + lane * 16;
      v16bf b = *(const v16bf*)bp;  // 32B-aligned: 2x global_load_b128 from L2
#pragma unroll
      for (int mt = 0; mt < MTILES; ++mt)
        acc[mt][i] = __builtin_amdgcn_wmma_f32_16x16x32_bf16(
            false, a[mt], false, b, (short)0, acc[mt][i], false, false);
    }
  }
}

// ---------------------------------------------------------------------------
// Fused: o2 = swish(BM2(swish(BM1(x*g1+nb1)+b1)*g3+nb3)+b2) + x
// One workgroup = 64 rows x one partition; both GEMMs + activations stay
// on-chip; residual x re-read from global in the final epilogue.
// ---------------------------------------------------------------------------
__global__ __launch_bounds__(256)
void bt_fused(const float* __restrict__ x,
              const bf16_t* __restrict__ wf,
              const float* __restrict__ bias1, const float* __restrict__ bias2,
              const float* __restrict__ gamma1, const float* __restrict__ beta1,
              const float* __restrict__ gamma3, const float* __restrict__ beta3,
              const float* __restrict__ gain1p, const float* __restrict__ nbias1p,
              const float* __restrict__ gain3p, const float* __restrict__ nbias3p,
              float* __restrict__ out) {
  __shared__ bf16_t Atile[ROWS][APAD];  // scaled bf16 input tile (GEMM1 A)
  __shared__ bf16_t O1t[ROWS][APAD];    // scaled bf16 hidden tile (GEMM2 A)

  const int p       = blockIdx.y;
  const int rowbase = blockIdx.x * ROWS;
  const int tid     = threadIdx.x;
  const int wave    = tid >> 5;
  const int lane    = tid & 31;
  const int lr      = lane & 15;
  const int hi      = lane >> 4;

  const float gain1 = gain1p[0], nb1 = nbias1p[0];
  const float gain3 = gain3p[0], nb3 = nbias3p[0];

  // ---- Stage X tile: (x*gain1+nbias1) as bf16 GEMM-A operand
  for (int it = tid; it < ROWS * (DDIM / 4); it += 256) {
    const int row = it >> 7;          // 128 float4 per row
    const int c4  = (it & 127) << 2;
    const float4 v = *(const float4*)&x[(size_t)(rowbase + row) * IN_SIZE + p * DDIM + c4];
    v4bf a;
    a.x = (bf16_t)(v.x * gain1 + nb1);
    a.y = (bf16_t)(v.y * gain1 + nb1);
    a.z = (bf16_t)(v.z * gain1 + nb1);
    a.w = (bf16_t)(v.w * gain1 + nb1);
    *(v4bf*)&Atile[row][c4] = a;
  }
  __syncthreads();

  // ---- GEMM1: Atile x W1[p]
  v8f acc[MTILES][4];
#pragma unroll
  for (int mt = 0; mt < MTILES; ++mt)
#pragma unroll
    for (int i = 0; i < 4; ++i) { v8f z = {}; acc[mt][i] = z; }

  const bf16_t* w1base = wf + (size_t)p * (DDIM * DDIM);
  gemm_tile(Atile, w1base, wave, lane, acc);

  // ---- Epilogue 1: +bias1, swish(gamma1,beta1), fold *gain3+nbias3, -> bf16 LDS
#pragma unroll
  for (int i = 0; i < 4; ++i) {
    const int col = wave * 64 + i * 16 + lr;
    const int c   = p * DDIM + col;
    const float bv = bias1[c], g = gamma1[c], bb = beta1[c];
#pragma unroll
    for (int mt = 0; mt < MTILES; ++mt) {
      const int rb = mt * 16 + hi * 8;
#pragma unroll
      for (int j = 0; j < 8; ++j) {
        const float o1 = pswish(acc[mt][i][j] + bv, g, bb);
        O1t[rb + j][col] = (bf16_t)(o1 * gain3 + nb3);
      }
    }
  }
  __syncthreads();

  // ---- GEMM2: O1t x W2[p]
#pragma unroll
  for (int mt = 0; mt < MTILES; ++mt)
#pragma unroll
    for (int i = 0; i < 4; ++i) { v8f z = {}; acc[mt][i] = z; }

  const bf16_t* w2base = wf + (size_t)(NPART + p) * (DDIM * DDIM);
  gemm_tile(O1t, w2base, wave, lane, acc);

  // ---- Epilogue 2: +bias2, swish(gamma3,beta3), + x residual (fp32), store
#pragma unroll
  for (int i = 0; i < 4; ++i) {
    const int col = wave * 64 + i * 16 + lr;
    const int c   = p * DDIM + col;
    const float bv = bias2[c], g = gamma3[c], bb = beta3[c];
#pragma unroll
    for (int mt = 0; mt < MTILES; ++mt) {
      const int rb = mt * 16 + hi * 8;
#pragma unroll
      for (int j = 0; j < 8; ++j) {
        const size_t gidx = (size_t)(rowbase + rb + j) * IN_SIZE + c;
        out[gidx] = pswish(acc[mt][i][j] + bv, g, bb) + x[gidx];
      }
    }
  }
}

extern "C" void kernel_launch(void* const* d_in, const int* in_sizes, int n_in,
                              void* d_out, int out_size, void* d_ws, size_t ws_size,
                              hipStream_t stream) {
  (void)in_sizes; (void)n_in; (void)out_size; (void)ws_size;
  const float* x     = (const float*)d_in[0];
  const float* w1    = (const float*)d_in[1];
  const float* b1    = (const float*)d_in[2];
  const float* w2    = (const float*)d_in[3];
  const float* b2    = (const float*)d_in[4];
  const float* g1    = (const float*)d_in[5];
  const float* be1   = (const float*)d_in[6];
  const float* g3    = (const float*)d_in[7];
  const float* be3   = (const float*)d_in[8];
  const float* gain1 = (const float*)d_in[9];
  const float* nb1   = (const float*)d_in[10];
  const float* gain3 = (const float*)d_in[11];
  const float* nb3   = (const float*)d_in[12];
  bf16_t* wfrag = (bf16_t*)d_ws;  // 8 MB: both weight tensors, bf16, frag layout

  const int total_w = 2 * NPART * DDIM * DDIM;
  bt_prep_weights<<<total_w / 256, 256, 0, stream>>>(w1, w2, wfrag);

  dim3 grid(B_ROWS / ROWS, NPART);
  bt_fused<<<grid, 256, 0, stream>>>(x, wfrag, b1, b2, g1, be1, g3, be3,
                                     gain1, nb1, gain3, nb3, (float*)d_out);
}
